// TGATModel_10350871184026
// MI455X (gfx1250) — matrix operand order repaired and checked by reading, not compiled, not run on hardware
//
#include <hip/hip_runtime.h>
#include <hip/hip_bf16.h>
#include <math.h>

#define N_NODES 100000
#define N_EDGES 800000
#define CIN     17
#define HID     128
#define NHEAD   8
#define DHEAD   16
#define TDIM    64
#define NBATCH  50000
#define BN_EPS  1e-5f

typedef __bf16 bf16;
typedef __attribute__((ext_vector_type(16))) __bf16 v16bf;
typedef __attribute__((ext_vector_type(8)))  __bf16 v8bf;
typedef __attribute__((ext_vector_type(8)))  float  v8f;

// ---------------- time encodings ----------------
__global__ void k_time_enc_f32(const float* __restrict__ t, const float* __restrict__ bf,
                               const float* __restrict__ ph, float* __restrict__ out, int n) {
    int i = blockIdx.x * blockDim.x + threadIdx.x;
    if (i >= n * TDIM) return;
    int r = i >> 6, j = i & 63;
    out[i] = cosf(t[r] * bf[j] + ph[j]);
}

__global__ void k_time_enc_bf16(const float* __restrict__ t, const float* __restrict__ bf,
                                const float* __restrict__ ph, bf16* __restrict__ out, int n) {
    int i = blockIdx.x * blockDim.x + threadIdx.x;
    if (i >= n * TDIM) return;
    int r = i >> 6, j = i & 63;
    out[i] = (bf16)cosf(t[r] * bf[j] + ph[j]);
}

// ---------------- weight prep: transpose, pad K, convert to bf16 ----------------
// W: [cin, nout] row-major f32  ->  Wt: [nout, Kpad] row-major bf16 (zero padded)
__global__ void k_prep_wt(const float* __restrict__ W, bf16* __restrict__ Wt,
                          int cin, int Kpad, int nout) {
    int i = blockIdx.x * blockDim.x + threadIdx.x;
    if (i >= nout * Kpad) return;
    int col = i / Kpad, kp = i % Kpad;
    float v = (kp < cin) ? W[(long)kp * nout + col] : 0.f;
    Wt[i] = (bf16)v;
}

// ---------------- concat [h | node_enc] -> bf16, zero-pad to Kpad ----------------
__global__ void k_concat_bf16(const float* __restrict__ h, const float* __restrict__ enc,
                              bf16* __restrict__ A, long total, int c, int Kpad) {
    long i = (long)blockIdx.x * blockDim.x + threadIdx.x;
    if (i >= total) return;
    long n = i / Kpad;
    int  j = (int)(i % Kpad);
    float v;
    if (j < c)              v = h[n * (long)c + j];
    else if (j < c + TDIM)  v = enc[n * (long)TDIM + (j - c)];
    else                    v = 0.f;
    A[i] = (bf16)v;
}

// ---------------- WMMA GEMM: out[M, NCT*16] = A[M,Kpad](bf16) @ Wt^T + bias --------------
// Wt stored as [Ncols, Kpad] (contiguous K per output column).
// Block = 128 threads (4 waves). Each wave owns ONE 16-row tile and sweeps all NCT column
// tiles: one A-fragment load feeds NCT back-to-back WMMAs (8 independent accumulator
// chains for NCT=8 -> hides XDL latency, no EXEC-mask games, A read exactly once).
template<int NCT>
__global__ void __launch_bounds__(128)
k_gemm_wmma(const bf16* __restrict__ A, const bf16* __restrict__ Wt,
            const float* __restrict__ bias, float* __restrict__ out,
            int M, int Kpad) {
    const int wid  = threadIdx.x >> 5;
    const int lane = threadIdx.x & 31;
    const int g    = lane >> 4;       // half-wave group
    const int n0   = lane & 15;
    const long R   = ((long)blockIdx.x * 4 + wid) * 16;   // this wave's row tile
    if (R >= M) return;               // wave-uniform: EXEC stays all-ones for live waves

    const int Ncols = NCT * 16;
    v8f acc[NCT];
    #pragma unroll
    for (int t = 0; t < NCT; ++t) acc[t] = (v8f){};

    const bf16* arow  = A  + (R + n0) * (long)Kpad;
    const bf16* bbase = Wt + (long)n0 * Kpad + g * 8;

    for (int kk = 0; kk < Kpad; kk += 32) {
        // A fragment (16-bit 16x32 layout): lane holds K[g*8..g*8+7] and K[16+g*8..16+g*8+7]
        union { v16bf v; v8bf h[2]; } af;
        af.h[0] = *(const v8bf*)(arow + kk + g * 8);
        af.h[1] = *(const v8bf*)(arow + kk + 16 + g * 8);
        #pragma unroll
        for (int t = 0; t < NCT; ++t) {
            union { v16bf v; v8bf h[2]; } bf_;
            bf_.h[0] = *(const v8bf*)(bbase + (long)t * 16 * Kpad + kk);
            bf_.h[1] = *(const v8bf*)(bbase + (long)t * 16 * Kpad + kk + 16);
            acc[t] = __builtin_amdgcn_wmma_f32_16x16x32_bf16(false, af.v, false, bf_.v,
                                                             (short)0, acc[t], false, false);
        }
    }

    #pragma unroll
    for (int t = 0; t < NCT; ++t) {
        int col = t * 16 + n0;
        float bv = bias ? bias[col] : 0.f;
        #pragma unroll
        for (int r = 0; r < 8; ++r)
            out[(R + g * 8 + r) * (long)Ncols + col] = acc[t][r] + bv;
    }
}

// ---------------- misc ----------------
__global__ void k_zero(float* __restrict__ p, long n) {
    long i = (long)blockIdx.x * blockDim.x + threadIdx.x;
    if (i < n) p[i] = 0.f;
}

// monotone float<->uint mapping for atomicMax on floats (handles negatives)
__device__ __forceinline__ unsigned fenc(float f) {
    unsigned u = __float_as_uint(f);
    return (u & 0x80000000u) ? ~u : (u | 0x80000000u);
}
__device__ __forceinline__ float fdec(unsigned u) {
    return (u & 0x80000000u) ? __uint_as_float(u & 0x7fffffffu) : __uint_as_float(~u);
}

// ---------------- edge attention ----------------
__global__ void __launch_bounds__(256)
k_edge_logits(const float* __restrict__ q, const float* __restrict__ k,
              const float* __restrict__ ep, const int* __restrict__ src,
              const int* __restrict__ dst, float* __restrict__ logits,
              unsigned* __restrict__ maxb) {
    long t = (long)blockIdx.x * blockDim.x + threadIdx.x;
    if (t >= (long)N_EDGES * NHEAD) return;
    int e = (int)(t >> 3), h = (int)(t & 7);
    int s = src[e], d = dst[e];
    const float4* qp = (const float4*)(q  + (long)d * HID + h * DHEAD);
    const float4* kp = (const float4*)(k  + (long)s * HID + h * DHEAD);
    const float4* xp = (const float4*)(ep + (long)e * HID + h * DHEAD);
    float acc = 0.f;
    #pragma unroll
    for (int i = 0; i < 4; ++i) {
        float4 a = qp[i], b = kp[i], c = xp[i];
        acc += a.x * (b.x + c.x) + a.y * (b.y + c.y) + a.z * (b.z + c.z) + a.w * (b.w + c.w);
    }
    float lg = acc * 0.25f;  // 1/sqrt(D_HEAD)
    logits[t] = lg;
    atomicMax(&maxb[(long)d * NHEAD + h], fenc(lg));
}

__global__ void __launch_bounds__(256)
k_edge_exp(float* __restrict__ logits, const int* __restrict__ dst,
           const unsigned* __restrict__ maxb, float* __restrict__ sumb) {
    long t = (long)blockIdx.x * blockDim.x + threadIdx.x;
    if (t >= (long)N_EDGES * NHEAD) return;
    int e = (int)(t >> 3), h = (int)(t & 7);
    int d = dst[e];
    float m = fdec(maxb[(long)d * NHEAD + h]);
    float ex = __expf(logits[t] - m);
    logits[t] = ex;
    atomicAdd(&sumb[(long)d * NHEAD + h], ex);
}

__global__ void __launch_bounds__(256)
k_edge_scatter(const float* __restrict__ v, const float* __restrict__ ep,
               const float* __restrict__ logits, const int* __restrict__ src,
               const int* __restrict__ dst, const float* __restrict__ sumb,
               float* __restrict__ outp) {
    long t = (long)blockIdx.x * blockDim.x + threadIdx.x;
    if (t >= (long)N_EDGES * NHEAD) return;
    int e = (int)(t >> 3), h = (int)(t & 7);
    int s = src[e], d = dst[e];
    float al = logits[t] / (sumb[(long)d * NHEAD + h] + 1e-16f);
    const float4* vp = (const float4*)(v  + (long)s * HID + h * DHEAD);
    const float4* xp = (const float4*)(ep + (long)e * HID + h * DHEAD);
    float* op = outp + (long)d * HID + h * DHEAD;
    #pragma unroll
    for (int i = 0; i < 4; ++i) {
        float4 a = vp[i], b = xp[i];
        atomicAdd(op + i * 4 + 0, (a.x + b.x) * al);
        atomicAdd(op + i * 4 + 1, (a.y + b.y) * al);
        atomicAdd(op + i * 4 + 2, (a.z + b.z) * al);
        atomicAdd(op + i * 4 + 3, (a.w + b.w) * al);
    }
}

// ---------------- batch norm ----------------
__global__ void k_bn_stats(const float* __restrict__ x, float* __restrict__ acc,
                           int rows, int C, int relu_first) {
    int c = threadIdx.x;          // blockDim.x == C
    float s = 0.f, s2 = 0.f;
    for (int r = blockIdx.x; r < rows; r += gridDim.x) {
        float v = x[(long)r * C + c];
        if (relu_first) v = fmaxf(v, 0.f);
        s += v; s2 += v * v;
    }
    atomicAdd(&acc[c], s);
    atomicAdd(&acc[C + c], s2);
}

__global__ void k_bn_finalize(const float* __restrict__ acc, const float* __restrict__ g,
                              const float* __restrict__ b, float* __restrict__ ss,
                              float inv, int C) {
    int c = threadIdx.x;
    if (c >= C) return;
    float mu  = acc[c] * inv;
    float var = acc[C + c] * inv - mu * mu;
    float rs  = rsqrtf(var + BN_EPS);
    float sc  = g[c] * rs;
    ss[c]     = sc;
    ss[C + c] = b[c] - mu * sc;
}

// mode 0: out = relu(x)*scale + shift   (layer BN: relu applied before norm)
// mode 1: out = relu(x*scale + shift)   (classifier: norm then relu)
__global__ void k_bn_apply(const float* __restrict__ x, const float* __restrict__ ss,
                           float* __restrict__ out, long total, int C, int mode) {
    long i = (long)blockIdx.x * blockDim.x + threadIdx.x;
    if (i >= total) return;
    int c = (int)(i % C);
    float v = x[i], r;
    if (mode == 0) r = fmaxf(v, 0.f) * ss[c] + ss[C + c];
    else           r = fmaxf(v * ss[c] + ss[C + c], 0.f);
    out[i] = r;
}

// ---------------- final 64 -> 1 projection ----------------
__global__ void k_clf_final(const float* __restrict__ t1, const float* __restrict__ W2,
                            const float* __restrict__ b2, float* __restrict__ out, int rows) {
    int r = blockIdx.x * blockDim.x + threadIdx.x;
    if (r >= rows) return;
    float acc = 0.f;
    #pragma unroll
    for (int j = 0; j < 64; ++j) acc += t1[(long)r * 64 + j] * W2[j];
    out[r] = acc + b2[0];
}

// ==================================================================
extern "C" void kernel_launch(void* const* d_in, const int* in_sizes, int n_in,
                              void* d_out, int out_size, void* d_ws, size_t ws_size,
                              hipStream_t stream) {
    (void)in_sizes; (void)n_in; (void)out_size; (void)ws_size;
    auto F = [&](int i) { return (const float*)d_in[i]; };

    const float* x     = F(0);
    const int*   eidx  = (const int*)d_in[1];
    const int*   src   = eidx;
    const int*   dst   = eidx + N_EDGES;
    const float* eattr = F(2);
    const float* ntime = F(3);
    const float* bfreq = F(5);
    const float* phase = F(6);

    // ---------- workspace carve ----------
    char* ws = (char*)d_ws; size_t off = 0;
    auto alloc = [&](size_t bytes) -> void* {
        void* p = ws + off; off += (bytes + 255) & ~(size_t)255; return p;
    };
    float*    node_enc = (float*)alloc((size_t)N_NODES * TDIM * 4);
    bf16*     eenc     = (bf16*) alloc((size_t)N_EDGES * TDIM * 2);
    bf16*     Abf      = (bf16*) alloc((size_t)N_NODES * 192 * 2);
    float*    qb       = (float*)alloc((size_t)N_NODES * HID * 4);
    float*    kb       = (float*)alloc((size_t)N_NODES * HID * 4);
    float*    vb       = (float*)alloc((size_t)N_NODES * HID * 4);
    float*    hconv    = (float*)alloc((size_t)N_NODES * HID * 4);
    float*    hcur     = (float*)alloc((size_t)N_NODES * HID * 4);
    float*    eproj    = (float*)alloc((size_t)N_EDGES * HID * 4);
    float*    logits   = (float*)alloc((size_t)N_EDGES * NHEAD * 4);
    unsigned* maxb     = (unsigned*)alloc((size_t)N_NODES * NHEAD * 4);
    float*    sumb     = (float*)alloc((size_t)N_NODES * NHEAD * 4);
    float*    bnacc    = (float*)alloc(2 * HID * 4);
    float*    bnss     = (float*)alloc(2 * HID * 4);
    float*    t0       = (float*)alloc((size_t)NBATCH * HID * 4);
    float*    t1b      = (float*)alloc((size_t)NBATCH * 64 * 4);
    bf16*     wtq      = (bf16*) alloc((size_t)HID * 192 * 2);
    bf16*     wtk      = (bf16*) alloc((size_t)HID * 192 * 2);
    bf16*     wtv      = (bf16*) alloc((size_t)HID * 192 * 2);
    bf16*     wts      = (bf16*) alloc((size_t)HID * 192 * 2);
    bf16*     wte      = (bf16*) alloc((size_t)HID * 64 * 2);

    // grid for the WMMA GEMM: 4 row tiles (waves) per block
    auto ggrid = [](int M) { return ((M / 16) + 3) / 4; };

    // ---------- time encodings ----------
    {
        long tot = (long)N_NODES * TDIM;
        k_time_enc_f32<<<(int)((tot + 255) / 256), 256, 0, stream>>>(ntime, bfreq, phase, node_enc, N_NODES);
    }
    {
        long tot = (long)N_EDGES * TDIM;
        k_time_enc_bf16<<<(int)((tot + 255) / 256), 256, 0, stream>>>(eattr, bfreq, phase, eenc, N_EDGES);
    }

    // ---------- GNN layers ----------
    const float* hprev = x;
    int cprev = CIN;
    for (int l = 0; l < 2; ++l) {
        const int base = 7 + l * 9;             // q.W q.b k.W k.b v.W v.b e.W skip.W skip.b
        const int cin  = cprev + TDIM;
        const int Kpad = (cin + 31) & ~31;      // 96 or 192

        int totw = HID * Kpad;
        k_prep_wt<<<(totw + 255) / 256, 256, 0, stream>>>(F(base + 0), wtq, cin, Kpad, HID);
        k_prep_wt<<<(totw + 255) / 256, 256, 0, stream>>>(F(base + 2), wtk, cin, Kpad, HID);
        k_prep_wt<<<(totw + 255) / 256, 256, 0, stream>>>(F(base + 4), wtv, cin, Kpad, HID);
        k_prep_wt<<<(totw + 255) / 256, 256, 0, stream>>>(F(base + 7), wts, cin, Kpad, HID);
        int tote = HID * TDIM;
        k_prep_wt<<<(tote + 255) / 256, 256, 0, stream>>>(F(base + 6), wte, TDIM, TDIM, HID);

        long totA = (long)N_NODES * Kpad;
        k_concat_bf16<<<(int)((totA + 255) / 256), 256, 0, stream>>>(hprev, node_enc, Abf, totA, cprev, Kpad);

        k_gemm_wmma<8><<<ggrid(N_NODES), 128, 0, stream>>>(Abf, wtq, F(base + 1), qb,    N_NODES, Kpad);
        k_gemm_wmma<8><<<ggrid(N_NODES), 128, 0, stream>>>(Abf, wtk, F(base + 3), kb,    N_NODES, Kpad);
        k_gemm_wmma<8><<<ggrid(N_NODES), 128, 0, stream>>>(Abf, wtv, F(base + 5), vb,    N_NODES, Kpad);
        k_gemm_wmma<8><<<ggrid(N_NODES), 128, 0, stream>>>(Abf, wts, F(base + 8), hconv, N_NODES, Kpad);
        k_gemm_wmma<8><<<ggrid(N_EDGES), 128, 0, stream>>>(eenc, wte, nullptr,    eproj, N_EDGES, TDIM);

        long nh = (long)N_NODES * NHEAD;
        k_zero<<<(int)((nh + 255) / 256), 256, 0, stream>>>((float*)maxb, nh);
        k_zero<<<(int)((nh + 255) / 256), 256, 0, stream>>>(sumb, nh);

        long te = (long)N_EDGES * NHEAD;
        int  eg = (int)((te + 255) / 256);
        k_edge_logits <<<eg, 256, 0, stream>>>(qb, kb, eproj, src, dst, logits, maxb);
        k_edge_exp    <<<eg, 256, 0, stream>>>(logits, dst, maxb, sumb);
        k_edge_scatter<<<eg, 256, 0, stream>>>(vb, eproj, logits, src, dst, sumb, hconv);

        k_zero<<<1, 256, 0, stream>>>(bnacc, 2 * HID);
        k_bn_stats<<<2048, HID, 0, stream>>>(hconv, bnacc, N_NODES, HID, 1);
        k_bn_finalize<<<1, HID, 0, stream>>>(bnacc, F(25 + l * 2), F(26 + l * 2), bnss,
                                             1.0f / (float)N_NODES, HID);
        long th = (long)N_NODES * HID;
        k_bn_apply<<<(int)((th + 255) / 256), 256, 0, stream>>>(hconv, bnss, hcur, th, HID, 0);

        hprev = hcur; cprev = HID;
    }

    // ---------- classifier head on first NBATCH rows ----------
    long tb = (long)NBATCH * HID;
    k_concat_bf16<<<(int)((tb + 255) / 256), 256, 0, stream>>>(hcur, node_enc, Abf, tb, HID, HID);
    k_prep_wt<<<(HID * HID + 255) / 256, 256, 0, stream>>>(F(29), wtq, HID, HID, HID);
    k_gemm_wmma<8><<<ggrid(NBATCH), 128, 0, stream>>>(Abf, wtq, F(30), t0, NBATCH, HID);

    k_zero<<<1, 256, 0, stream>>>(bnacc, 2 * HID);
    k_bn_stats<<<2048, HID, 0, stream>>>(t0, bnacc, NBATCH, HID, 0);
    k_bn_finalize<<<1, HID, 0, stream>>>(bnacc, F(35), F(36), bnss, 1.0f / (float)NBATCH, HID);
    k_bn_apply<<<(int)((tb + 255) / 256), 256, 0, stream>>>(t0, bnss, t0, tb, HID, 1);

    k_concat_bf16<<<(int)((tb + 255) / 256), 256, 0, stream>>>(t0, node_enc, Abf, tb, HID, HID);
    k_prep_wt<<<(64 * HID + 255) / 256, 256, 0, stream>>>(F(31), wtk, HID, HID, 64);
    k_gemm_wmma<4><<<ggrid(NBATCH), 128, 0, stream>>>(Abf, wtk, F(32), t1b, NBATCH, HID);

    k_zero<<<1, 256, 0, stream>>>(bnacc, 2 * 64);
    k_bn_stats<<<2048, 64, 0, stream>>>(t1b, bnacc, NBATCH, 64, 0);
    k_bn_finalize<<<1, 64, 0, stream>>>(bnacc, F(37), F(38), bnss, 1.0f / (float)NBATCH, 64);
    long tt = (long)NBATCH * 64;
    k_bn_apply<<<(int)((tt + 255) / 256), 256, 0, stream>>>(t1b, bnss, t1b, tt, 64, 1);

    k_clf_final<<<(NBATCH + 255) / 256, 256, 0, stream>>>(t1b, F(33), F(34), (float*)d_out, NBATCH);
}
